// UserItemAggregator_73461120631292
// MI455X (gfx1250) — matrix operand order repaired and checked by reading, not compiled, not run on hardware
//
#include <hip/hip_runtime.h>
#include <cstddef>

// ---------------------------------------------------------------------------
// Types
// ---------------------------------------------------------------------------
typedef unsigned short ushort_t;
typedef unsigned int   uint32;
typedef unsigned long long uint64;

typedef __attribute__((ext_vector_type(16))) __bf16  v16bf;
typedef __attribute__((ext_vector_type(8)))  float   v8f;
typedef __attribute__((ext_vector_type(4)))  uint32  u32x4;
typedef __attribute__((ext_vector_type(4)))  float   f32x4;

union FragAB { v16bf v; u32x4 q[2]; };

// Native bf16 converts (gfx1250 has HW BF16 cvt ops)
static __device__ __forceinline__ ushort_t f2bf(float f) {
  __bf16 b = (__bf16)f;
  ushort_t r; __builtin_memcpy(&r, &b, 2); return r;
}
static __device__ __forceinline__ float bf2f(ushort_t s) {
  __bf16 b; __builtin_memcpy(&b, &s, 2); return (float)b;
}
static __device__ __forceinline__ uint32 pk2bf(float lo, float hi) {
  return (uint32)f2bf(lo) | ((uint32)f2bf(hi) << 16);
}

// ---------------------------------------------------------------------------
// CDNA5 async copy: global -> LDS (16B per op), tracked by ASYNCcnt
// ---------------------------------------------------------------------------
static __device__ __forceinline__ void async_b128_to_lds(void* lds_dst,
                                                         const void* gsrc) {
  uint32 lds_off = (uint32)(size_t)lds_dst;   // flat addr low 32 = LDS byte addr
  uint64 gaddr   = (uint64)(size_t)gsrc;
  asm volatile("global_load_async_to_lds_b128 %0, %1, off"
               :: "v"(lds_off), "v"(gaddr) : "memory");
}
static __device__ __forceinline__ void wait_async_all() {
  asm volatile("s_wait_asynccnt 0x0" ::: "memory");
}

// ---------------------------------------------------------------------------
// Problem constants (from reference)
// ---------------------------------------------------------------------------
#define NN   8192
#define DEG  64
#define DD   128

// ---------------------------------------------------------------------------
// Weight prep: fp32 [K][N] -> bf16 transposed [N][K] in workspace
// ---------------------------------------------------------------------------
__global__ void prep_weights_kernel(
    const float* __restrict__ w1,  const float* __restrict__ w2,
    const float* __restrict__ wa1, const float* __restrict__ wa2,
    const float* __restrict__ wa3,
    ushort_t* __restrict__ wt1,  ushort_t* __restrict__ wt2,
    ushort_t* __restrict__ wta1, ushort_t* __restrict__ wta2,
    ushort_t* __restrict__ wa3v)
{
  int gid = blockIdx.x * 256 + threadIdx.x;
  if (gid < 32768) {                       // w1: [256,128] -> wt1 [128][256]
    int n = gid & 127, k = gid >> 7;
    wt1[n * 256 + k] = f2bf(w1[k * 128 + n]);
  } else if (gid < 49152) {                // w2: [128,128] -> wt2 [128][128]
    int t = gid - 32768; int n = t & 127, k = t >> 7;
    wt2[n * 128 + k] = f2bf(w2[k * 128 + n]);
  } else if (gid < 81920) {                // wa1: [256,128] -> wta1 [128][256]
    int t = gid - 49152; int n = t & 127, k = t >> 7;
    wta1[n * 256 + k] = f2bf(wa1[k * 128 + n]);
  } else if (gid < 98304) {                // wa2: [128,128] -> wta2 [128][128]
    int t = gid - 81920; int n = t & 127, k = t >> 7;
    wta2[n * 128 + k] = f2bf(wa2[k * 128 + n]);
  } else if (gid < 98432) {                // wa3: [128,1] -> bf16 vector
    int t = gid - 98304;
    wa3v[t] = f2bf(wa3[t]);
  }
}

// ---------------------------------------------------------------------------
// Issue one weight chunk sW[n][32] (n=0..127, 8 KB) as 512 async b128 ops
// (2 per thread).
// ---------------------------------------------------------------------------
static __device__ __forceinline__ void issue_wchunk(
    const ushort_t* __restrict__ Wg, int wStride, int kb,
    ushort_t* sWbuf, int tid)
{
  #pragma unroll
  for (int i = 0; i < 2; ++i) {
    int e = tid + (i << 8);          // 0..511
    int n = e >> 2, c = e & 3;
    async_b128_to_lds(&sWbuf[(n << 5) + (c << 3)],
                      &Wg[n * wStride + kb + (c << 3)]);
  }
}

// ---------------------------------------------------------------------------
// One GEMM stage: D[64][128] = relu( A[64][K] @ W[K][128] + bias )
// 8 waves: wave w -> m-tile (w&3), n-block ((w>>2)*64, 4 n-tiles).
// Weights streamed via double-buffered async-to-LDS chunks (K=32 each).
// If P != nullptr, columns K>=128 of A come from the broadcast row P (concat).
// In-place (Dst aliasing A) is safe: accumulators live in registers until the
// post-loop barrier.
// ---------------------------------------------------------------------------
static __device__ __forceinline__ void gemm_stage(
    const ushort_t* __restrict__ A, int aStride,
    const ushort_t* __restrict__ P,
    int K,
    const ushort_t* __restrict__ Wg, int wStride,
    const float*    __restrict__ bias,
    ushort_t* __restrict__ Dst, int dStride,
    ushort_t* sW0, ushort_t* sW1,
    int tid)
{
  const int lane = tid & 31;
  const int wave = tid >> 5;
  const int mt   = wave & 3;            // m-tile index 0..3
  const int nb   = (wave >> 2) << 6;    // n base: 0 or 64
  const int g    = lane >> 4;           // lane group 0/1
  const int lm   = lane & 15;
  const int m    = (mt << 4) + lm;      // A row for this lane

  v8f acc[4];
  acc[0] = v8f{}; acc[1] = v8f{}; acc[2] = v8f{}; acc[3] = v8f{};

  const int ksteps = K >> 5;

  // Prologue: kick off chunk 0. Safe without a barrier: every thread passed
  // the previous stage's post-loop barrier, i.e. all sW reads are retired.
  issue_wchunk(Wg, wStride, 0, sW0, tid);

  for (int ks = 0; ks < ksteps; ++ks) {
    const int kb = ks << 5;
    ushort_t* sWcur = (ks & 1) ? sW1 : sW0;
    ushort_t* sWnxt = (ks & 1) ? sW0 : sW1;

    wait_async_all();     // this thread's chunk-ks ops have landed in LDS
    __syncthreads();      // everyone's chunk ks visible; everyone done
                          // reading chunk ks-1 (consumed by its WMMAs)
    if (ks + 1 < ksteps)  // prefetch next chunk while we compute
      issue_wchunk(Wg, wStride, kb + 32, sWnxt, tid);

    // A fragment (ISA 16-bit A 16x32 layout: two contiguous 8-elem chunks)
    FragAB a;
    {
      const ushort_t* row; int kl;
      if (P != nullptr && kb >= 128) { row = P; kl = kb - 128; }
      else                           { row = A + m * aStride; kl = kb; }
      a.q[0] = *(const u32x4*)&row[kl + (g << 3)];
      a.q[1] = *(const u32x4*)&row[kl + 16 + (g << 3)];
    }

    #pragma unroll
    for (int j = 0; j < 4; ++j) {
      // B fragment (lane = n, same K layout as A: two 8-elem chunks at 8g and 16+8g)
      FragAB b;
      const int n = nb + (j << 4) + lm;
      b.q[0] = *(const u32x4*)&sWcur[(n << 5) + (g << 3)];
      b.q[1] = *(const u32x4*)&sWcur[(n << 5) + 16 + (g << 3)];
      acc[j] = __builtin_amdgcn_wmma_f32_16x16x32_bf16(
          false, a.v, false, b.v, (short)0, acc[j], false, false);
    }
  }

  __syncthreads();   // all waves done reading A before Dst may alias it
  #pragma unroll
  for (int j = 0; j < 4; ++j) {
    const int n = nb + (j << 4) + lm;
    const float bn = bias[n];
    v8f av = acc[j];
    #pragma unroll
    for (int v = 0; v < 8; ++v) {
      const int mr = (mt << 4) + (g << 3) + v;     // C/D layout row
      float x = av[v] + bn;
      x = x > 0.0f ? x : 0.0f;
      Dst[mr * dStride + n] = f2bf(x);
    }
  }
}

// ---------------------------------------------------------------------------
// Main kernel: one block (256 threads = 8 waves) per node
// ---------------------------------------------------------------------------
__global__ __launch_bounds__(256) void UserItemAggregator_kernel(
    const int*   __restrict__ nodes,
    const int*   __restrict__ adj,
    const int*   __restrict__ ratings,
    const float* __restrict__ user_emb,
    const float* __restrict__ item_emb,
    const float* __restrict__ rating_emb,
    const float* __restrict__ b1,  const float* __restrict__ b2,
    const float* __restrict__ ba1, const float* __restrict__ ba2,
    const float* __restrict__ ba3,
    const ushort_t* __restrict__ wt1,
    const ushort_t* __restrict__ wt2,
    const ushort_t* __restrict__ wta1,
    const ushort_t* __restrict__ wta2,
    const ushort_t* __restrict__ wa3v,
    float* __restrict__ out)
{
  __shared__ alignas(16) ushort_t sAct[DEG * 256];   // 32 KB: X0, then planes
  __shared__ alignas(16) ushort_t sW[2][DD * 32];    // 2 x 8 KB weight chunks
  __shared__ alignas(16) ushort_t sP[DD];
  __shared__ alignas(16) ushort_t sWa3[DD];
  __shared__ int   sAdj[DEG];
  __shared__ int   sRat[DEG];
  __shared__ float sLogit[DEG];
  __shared__ float sAttn[DEG];

  const int tid  = threadIdx.x;
  const int node = blockIdx.x;

  if (tid < DEG) {
    sAdj[tid] = adj[(size_t)node * DEG + tid];
    sRat[tid] = ratings[(size_t)node * DEG + tid];
  }
  if (tid < DD) sWa3[tid] = wa3v[tid];
  __syncthreads();

  // Gather: X0[row][0:128) = item_emb[adj], X0[row][128:256) = rating_emb[rat]
  for (int i = tid; i < DEG * 32; i += 256) {
    int row = i >> 5, seg = i & 31;
    const f32x4 ve = *(const f32x4*)&item_emb[(size_t)sAdj[row] * DD + seg * 4];
    int o = row * 256 + seg * 4;
    *(uint32*)&sAct[o]     = pk2bf(ve.x, ve.y);
    *(uint32*)&sAct[o + 2] = pk2bf(ve.z, ve.w);
    const f32x4 vr = *(const f32x4*)&rating_emb[(size_t)sRat[row] * DD + seg * 4];
    *(uint32*)&sAct[o + 128]     = pk2bf(vr.x, vr.y);
    *(uint32*)&sAct[o + 128 + 2] = pk2bf(vr.z, vr.w);
  }
  if (tid < 32) {
    const f32x4 vp = *(const f32x4*)&user_emb[(size_t)nodes[node] * DD + tid * 4];
    *(uint32*)&sP[tid * 4]     = pk2bf(vp.x, vp.y);
    *(uint32*)&sP[tid * 4 + 2] = pk2bf(vp.z, vp.w);
  }
  // (visibility handled by the barrier inside gemm_stage's first k-step)

  ushort_t* plane0 = sAct;              // [64][128]
  ushort_t* plane1 = sAct + DEG * DD;   // [64][128]

  // H  = relu(X0 @ W1 + b1)            -> plane0
  gemm_stage(sAct,   256, nullptr, 256, wt1,  256, b1,  plane0, DD, sW[0], sW[1], tid);
  // X  = relu(H @ W2 + b2)             -> plane1  (kept until the end)
  gemm_stage(plane0, DD,  nullptr, 128, wt2,  DD,  b2,  plane1, DD, sW[0], sW[1], tid);
  // A1 = relu(cat(X, P) @ Wa1 + ba1)   -> plane0
  gemm_stage(plane1, DD,  sP,      256, wta1, 256, ba1, plane0, DD, sW[0], sW[1], tid);
  // A2 = relu(A1 @ Wa2 + ba2)          -> plane0 (in-place; regs until barrier)
  gemm_stage(plane0, DD,  nullptr, 128, wta2, DD,  ba2, plane0, DD, sW[0], sW[1], tid);

  __syncthreads();
  // logits[k] = A2[k] . wa3 + ba3
  if (tid < DEG) {
    float acc = ba3[0];
    #pragma unroll 4
    for (int d = 0; d < DD; ++d) acc += bf2f(plane0[tid * DD + d]) * bf2f(sWa3[d]);
    sLogit[tid] = acc;
  }
  __syncthreads();
  // softmax over 64 neighbors (redundant per-thread scan; trivial cost)
  if (tid < DEG) {
    float mx = -3.4e38f;
    for (int k = 0; k < DEG; ++k) mx = fmaxf(mx, sLogit[k]);
    float s = 0.0f;
    for (int k = 0; k < DEG; ++k) s += __expf(sLogit[k] - mx);
    sAttn[tid] = __expf(sLogit[tid] - mx) / s;
  }
  __syncthreads();
  // out[node][d] = sum_k attn[k] * X[k][d]
  if (tid < DD) {
    float acc = 0.0f;
    for (int k = 0; k < DEG; ++k) acc += sAttn[k] * bf2f(plane1[k * DD + tid]);
    out[(size_t)node * DD + tid] = acc;
  }
}

// ---------------------------------------------------------------------------
// Launcher
// ---------------------------------------------------------------------------
extern "C" void kernel_launch(void* const* d_in, const int* in_sizes, int n_in,
                              void* d_out, int out_size, void* d_ws, size_t ws_size,
                              hipStream_t stream) {
  const int*   nodes      = (const int*)d_in[0];
  const int*   adj        = (const int*)d_in[1];
  const int*   ratings    = (const int*)d_in[2];
  const float* user_emb   = (const float*)d_in[3];
  const float* item_emb   = (const float*)d_in[4];
  const float* rating_emb = (const float*)d_in[5];
  const float* w1  = (const float*)d_in[6];
  const float* b1  = (const float*)d_in[7];
  const float* w2  = (const float*)d_in[8];
  const float* b2  = (const float*)d_in[9];
  const float* wa1 = (const float*)d_in[10];
  const float* ba1 = (const float*)d_in[11];
  const float* wa2 = (const float*)d_in[12];
  const float* ba2 = (const float*)d_in[13];
  const float* wa3 = (const float*)d_in[14];
  const float* ba3 = (const float*)d_in[15];

  char* ws = (char*)d_ws;
  ushort_t* wt1  = (ushort_t*)(ws);
  ushort_t* wt2  = (ushort_t*)(ws + 65536);
  ushort_t* wta1 = (ushort_t*)(ws + 65536 + 32768);
  ushort_t* wta2 = (ushort_t*)(ws + 65536 + 32768 + 65536);
  ushort_t* wa3v = (ushort_t*)(ws + 65536 + 32768 + 65536 + 32768);

  prep_weights_kernel<<<385, 256, 0, stream>>>(w1, w2, wa1, wa2, wa3,
                                               wt1, wt2, wta1, wta2, wa3v);
  UserItemAggregator_kernel<<<NN, 256, 0, stream>>>(
      nodes, adj, ratings, user_emb, item_emb, rating_emb,
      b1, b2, ba1, ba2, ba3,
      wt1, wt2, wta1, wta2, wa3v, (float*)d_out);
}
